// V_theta_69209103007966
// MI455X (gfx1250) — matrix operand order repaired and checked by reading, not compiled, not run on hardware
//
#include <hip/hip_runtime.h>
#include <cmath>
#include <cstdint>
#include <vector>

typedef __attribute__((ext_vector_type(16))) _Float16 v16h;
typedef __attribute__((ext_vector_type(8)))  float    v8f;

#define N_NODES 800
#define N_EDGES 12000
#define N_BONDS 800
#define NL 5                 // l = 0..4
#define W3J_STRIDE 729       // 9*9*9 per (l1,l2,l3) slot
#define MAXPATH 1024
#define MAXIR 96

// ---- optional CDNA5 async global->LDS staging (guarded; falls back to plain stores) ----
#if __has_builtin(__builtin_amdgcn_global_load_async_to_lds_b32) && __has_builtin(__builtin_amdgcn_s_wait_asynccnt)
#define HAVE_ASYNC 1
#else
#define HAVE_ASYNC 0
#endif

#if HAVE_ASYNC
typedef __attribute__((address_space(1))) int gas_int;
typedef __attribute__((address_space(3))) int las_int;
__device__ inline void stage_f32(const float* g, float* l){
  __builtin_amdgcn_global_load_async_to_lds_b32(
      (gas_int*)(unsigned long long)(uintptr_t)g,
      (las_int*)(unsigned int)(uintptr_t)l, 0, 0);
}
__device__ inline void stage_fence(){ __builtin_amdgcn_s_wait_asynccnt(0); }
#else
__device__ inline void stage_f32(const float* g, float* l){ *l = *g; }
__device__ inline void stage_fence(){}
#endif

// ---------------- irreps tables (mul, l, parity) — mirror of reference ----------------
static constexpr int IR_IN_[1][3]  = {{2,0,1}};
static constexpr int IR_SH_[3][3]  = {{1,0,1},{1,1,-1},{1,2,1}};
static constexpr int IR_M1_[3][3]  = {{8,0,1},{8,1,-1},{8,2,1}};
static constexpr int IR_M2_[6][3]  = {{8,0,1},{8,0,-1},{8,1,1},{8,1,-1},{8,2,1},{8,2,-1}};

#define HH_HALF {4,0,1},{2,1,-1},{2,1,-1},{1,0,1},{1,1,1},{1,2,1}
static constexpr int IR_HH_[12][3] = { HH_HALF, HH_HALF };

#define CC_HALF \
  {9,0,1},{6,1,-1},{3,2,1}, \
  {6,1,-1}, \
  {1,0,1},{1,1,1},{1,2,1}, {1,0,1},{1,1,1},{1,2,1}, {1,0,1},{1,1,1},{1,2,1}, {1,0,1},{1,1,1},{1,2,1}, \
  {1,1,-1},{1,2,-1},{1,3,-1}, {1,1,-1},{1,2,-1},{1,3,-1}, \
  {3,2,1}, \
  {1,1,-1},{1,2,-1},{1,3,-1}, {1,1,-1},{1,2,-1},{1,3,-1}, \
  {1,0,1},{1,1,1},{1,2,1},{1,3,1},{1,4,1}
static constexpr int IR_CC_[68][3] = { CC_HALF, CC_HALF };

#define CH_HALF \
  {6,0,1},{3,1,-1}, \
  {4,1,-1}, {1,0,1},{1,1,1},{1,2,1}, {1,0,1},{1,1,1},{1,2,1}, \
  {2,2,1}, {1,1,-1},{1,2,-1},{1,3,-1}
static constexpr int IR_CH_[26][3] = { CH_HALF, CH_HALF };

// ---------------- tensor-product spec (shared host/device builder) ----------------
struct TPPath { int o1,m1,l1,o2,m2,l2,o3,m3,l3,woff; float invfan; int i3; };
struct TPSpec { int npaths, numel, d1, d2, d3, wmax, _pad[2]; TPPath p[MAXPATH]; };

__host__ __device__ inline void build_tp(const int (*ir1)[3], int n1,
                                         const int (*ir2)[3], int n2,
                                         const int (*ir3)[3], int n3,
                                         TPSpec* s) {
  int o1[MAXIR], o2[MAXIR], o3[MAXIR];
  float fan[MAXIR];
  int d = 0;
  for (int i = 0; i < n1; i++) { o1[i] = d; d += ir1[i][0] * (2*ir1[i][1]+1); } s->d1 = d;
  d = 0;
  for (int i = 0; i < n2; i++) { o2[i] = d; d += ir2[i][0] * (2*ir2[i][1]+1); } s->d2 = d;
  d = 0;
  for (int i = 0; i < n3; i++) { o3[i] = d; fan[i] = 0.f; d += ir3[i][0] * (2*ir3[i][1]+1); } s->d3 = d;
  int np = 0, off = 0, wmax = 0;
  for (int i1 = 0; i1 < n1; i1++)
    for (int i2 = 0; i2 < n2; i2++)
      for (int i3 = 0; i3 < n3; i3++) {
        int l1 = ir1[i1][1], l2 = ir2[i2][1], l3 = ir3[i3][1];
        if (ir1[i1][2] * ir2[i2][2] != ir3[i3][2]) continue;
        int lo = l1 > l2 ? l1 - l2 : l2 - l1;
        if (l3 < lo || l3 > l1 + l2) continue;
        int wsz = ir1[i1][0] * ir2[i2][0] * ir3[i3][0];
        if (np < MAXPATH) {
          TPPath& P = s->p[np];
          P.o1 = o1[i1]; P.m1 = ir1[i1][0]; P.l1 = l1;
          P.o2 = o2[i2]; P.m2 = ir2[i2][0]; P.l2 = l2;
          P.o3 = o3[i3]; P.m3 = ir3[i3][0]; P.l3 = l3;
          P.woff = off; P.invfan = 0.f; P.i3 = i3;
          np++;
        }
        off += wsz;
        if (wsz > wmax) wmax = wsz;
        fan[i3] += (float)(ir1[i1][0] * ir2[i2][0]);
      }
  s->npaths = np; s->numel = off; s->wmax = (wmax + 15) & ~15;
  for (int k = 0; k < np; k++) s->p[k].invfan = 1.0f / sqrtf(fan[s->p[k].i3]);
}

// ---------------- workspace layout ----------------
constexpr int    W3J_FLOATS = NL*NL*NL*W3J_STRIDE;                       // 91125
constexpr size_t W3J_BYTES  = (size_t)W3J_FLOATS * 4;
constexpr size_t SPEC_OFF   = (W3J_BYTES + 255) & ~(size_t)255;
constexpr size_t SPEC_BYTES = 6 * sizeof(TPSpec);
constexpr size_t BUF_OFF    = (SPEC_OFF + SPEC_BYTES + 255) & ~(size_t)255;

// ---------------- device Wigner-3j generation (deterministic, per call) ----------------
struct cplx { double re, im; };
__device__ inline cplx cmul(cplx a, cplx b){ return {a.re*b.re - a.im*b.im, a.re*b.im + a.im*b.re}; }

__device__ double dfac(int n){ double r = 1.0; for (int i = 2; i <= n; i++) r *= (double)i; return r; }

__device__ double su2_cg_coeff(int j1,int m1,int j2,int m2,int j3,int m3){
  if (m3 != m1 + m2) return 0.0;
  int lo = j1 > j2 ? j1 - j2 : j2 - j1;
  if (j3 < lo || j3 > j1 + j2) return 0.0;
  double pref = (2.0*j3 + 1.0) * dfac(j3+j1-j2) * dfac(j3-j1+j2) * dfac(j1+j2-j3) / dfac(j1+j2+j3+1);
  pref *= dfac(j3+m3)*dfac(j3-m3)*dfac(j1-m1)*dfac(j1+m1)*dfac(j2-m2)*dfac(j2+m2);
  double s = 0.0;
  for (int k = 0; k <= j1+j2-j3; k++) {
    int d0=k, d1=j1+j2-j3-k, d2=j1-m1-k, d3=j2+m2-k, d4=j3-j2+m1+k, d5=j3-j1-m2+k;
    if (d0<0||d1<0||d2<0||d3<0||d4<0||d5<0) continue;
    double t = 1.0/(dfac(d0)*dfac(d1)*dfac(d2)*dfac(d3)*dfac(d4)*dfac(d5));
    s += (k & 1) ? -t : t;
  }
  return sqrt(pref) * s;
}

__device__ void q_r2c(int l, cplx q[9][9]){
  for (int a = 0; a < 9; a++) for (int b = 0; b < 9; b++) q[a][b] = {0.0, 0.0};
  double is2 = 1.0 / sqrt(2.0);
  for (int m = -l; m < 0; m++) { q[l+m][l-m] = {is2, 0.0}; q[l+m][l+m] = {0.0, -is2}; }
  q[l][l] = {1.0, 0.0};
  for (int m = 1; m <= l; m++) {
    double sg = (m & 1) ? -1.0 : 1.0;
    q[l+m][l+m] = {sg*is2, 0.0};
    q[l+m][l-m] = {0.0, sg*is2};
  }
  cplx ph;
  switch (l & 3) { case 0: ph = {1,0}; break; case 1: ph = {0,-1}; break;
                   case 2: ph = {-1,0}; break; default: ph = {0,1}; }
  int n = 2*l + 1;
  for (int a = 0; a < n; a++) for (int b = 0; b < n; b++) q[a][b] = cmul(ph, q[a][b]);
}

__device__ void compute_w3j(int l1, int l2, int l3, float* out /*729 slot*/){
  int n1 = 2*l1+1, n2 = 2*l2+1, n3 = 2*l3+1;
  cplx Q1[9][9], Q2[9][9], Q3[9][9];
  q_r2c(l1, Q1); q_r2c(l2, Q2); q_r2c(l3, Q3);
  for (int j = 0; j < n1; j++)
    for (int lc = 0; lc < n2; lc++)
      for (int m = 0; m < n3; m++) {
        double ar = 0.0;
        for (int i = 0; i < n1; i++) {
          int m1 = i - l1;
          for (int k = 0; k < n2; k++) {
            int m2 = k - l2, m3 = m1 + m2;
            if (m3 < -l3 || m3 > l3) continue;
            double cg = su2_cg_coeff(l1, m1, l2, m2, l3, m3);
            if (cg == 0.0) continue;
            cplx t = cmul(Q1[i][j], Q2[k][lc]);
            t = cmul(t, Q3[m][l3 + m3]);
            ar += t.re * cg;
          }
        }
        out[(j*9 + lc)*9 + m] = (float)ar;
      }
}

__global__ void init_kernel(unsigned char* ws){
  float* w3 = (float*)ws;
  int tid = threadIdx.x;
  for (int i = tid; i < W3J_FLOATS; i += blockDim.x) w3[i] = 0.f;
  __syncthreads();
  TPSpec* sp = (TPSpec*)(ws + SPEC_OFF);
  if (tid == 0) build_tp(IR_IN_, 1, IR_SH_, 3, IR_M1_, 3, &sp[0]);   // TP1
  if (tid == 1) build_tp(IR_M1_, 3, IR_SH_, 3, IR_M2_, 6, &sp[1]);   // TP2
  if (tid == 2) build_tp(IR_M2_, 6, IR_M2_, 6, IR_M2_, 6, &sp[2]);   // TP_BOND
  if (tid == 3) build_tp(IR_M2_, 6, IR_SH_, 3, IR_HH_, 12, &sp[3]);  // TP_HH / TP_H
  if (tid == 4) build_tp(IR_M2_, 6, IR_SH_, 3, IR_CC_, 68, &sp[4]);  // TP_CC / TP_C
  if (tid == 5) build_tp(IR_M2_, 6, IR_SH_, 3, IR_CH_, 26, &sp[5]);  // TP_CH
  for (int t = tid; t < NL*NL*NL; t += blockDim.x) {
    int l1 = t/25, l2 = (t/5)%5, l3 = t%5;
    int lo = l1 > l2 ? l1 - l2 : l2 - l1;
    if (l3 < lo || l3 > l1 + l2) continue;
    compute_w3j(l1, l2, l3, w3 + (size_t)t * W3J_STRIDE);
  }
}

// ---------------- final-layer weight cache: W[16][numel] f32 -> Wt[numel][16] f16 (x0.25) ----------------
__global__ void prep_wt_kernel(const float* __restrict__ W, int numel, _Float16* __restrict__ out){
  int i = blockIdx.x*blockDim.x + threadIdx.x;
  if (i >= numel*16) return;
  int col = i >> 4, j = i & 15;
  out[(size_t)col*16 + j] = (_Float16)(W[(size_t)j*numel + col] * 0.25f);
}

// ---------------- WMMA MLP hidden-ladder: emb -> relu(W0) -> [relu(16x16 WMMA)] x nextra ----------------
__device__ inline float relu_(float x){ return x > 0.f ? x : 0.f; }

__global__ __launch_bounds__(32)
void mlp_hidden_kernel(const float* __restrict__ emb,
                       const float* __restrict__ W0,
                       const float* __restrict__ W1,
                       const float* __restrict__ W2,
                       int nextra, float* __restrict__ hout){
  __shared__ float hl[16][17];
  int lane = threadIdx.x & 31;
  int r    = lane & 15;
  int hi   = lane >> 4;
  int e0   = blockIdx.x * 16;
  float ev = emb[e0 + r];
  for (int j = 0; j < 8; j++) { int c = hi*8 + j; hl[r][c] = relu_(ev * W0[c]); }
  __syncthreads();
  const float* Ws[2] = {W1, W2};
  for (int layer = 0; layer < nextra; ++layer) {
    const float* W = Ws[layer];
    v16h a = {};
    int kb = hi ? 8 : 0;
    for (int j = 0; j < 16; j++) a[j] = hi ? (_Float16)0.f : (_Float16)hl[r][j];
    v16h b = {};
    if (hi == 0) { for (int j = 0; j < 16; j++) b[j] = (_Float16)(W[j*16 + r] * 0.25f); }
    else         { for (int j = 0; j < 16; j++) b[j] = (_Float16)0.f; }
    v8f c = {};
    c = __builtin_amdgcn_wmma_f32_16x16x32_f16(false, a, false, b, (short)0, c, false, false);
    __syncthreads();
    for (int rr = 0; rr < 8; rr++) hl[rr + hi*8][r] = relu_(c[rr]);
    __syncthreads();
  }
  for (int j = 0; j < 8; j++) { int c = hi*8 + j; hout[(size_t)(e0 + r)*16 + c] = hl[r][c]; }
}

// ---------------- fused gather + TP + (scatter-add | store), 16 edges / block ----------------
// Per path: weight block w[16 edges][wsz] = h @ Wt[woff:woff+wsz] is materialized into LDS
// via v_wmma_f32_16x16x32_f16 tiles (B = contiguous 32B v16h rows of the f16 weight cache),
// then 256 threads run the Wigner contraction out of LDS.
__global__ __launch_bounds__(256)
void tp16_kernel(const float* __restrict__ x1, const int* __restrict__ idx1,
                 const float* __restrict__ x2, const int* __restrict__ idx2,
                 const float* __restrict__ h,  const _Float16* __restrict__ Wt,
                 float* __restrict__ yout, const int* __restrict__ dstidx, float outscale,
                 const TPSpec* __restrict__ spec, const float* __restrict__ w3j){
  extern __shared__ float lds[];
  const int t    = threadIdx.x;
  const int lane = t & 31;
  const int wv   = t >> 5;          // wave id, 0..7
  const int e0   = blockIdx.x * 16;
  const int d1 = spec->d1, d2 = spec->d2, d3 = spec->d3, wmax = spec->wmax;

  float* sx1 = lds;                  // 16*d1
  float* sx2 = sx1 + 16*d1;          // 16*d2
  float* shh = sx2 + 16*d2;          // 16*16
  float* sy  = shh + 256;            // 16*d3
  float* sw  = sy  + 16*d3;          // 16*wmax

  for (int i = t; i < 16*d1; i += 256) {
    int e = i / d1, c = i % d1;
    int row = idx1 ? idx1[e0 + e] : (e0 + e);
    stage_f32(&x1[(size_t)row*d1 + c], &sx1[i]);
  }
  for (int i = t; i < 16*d2; i += 256) {
    int e = i / d2, c = i % d2;
    int row = idx2 ? idx2[e0 + e] : (e0 + e);
    stage_f32(&x2[(size_t)row*d2 + c], &sx2[i]);
  }
  for (int i = t; i < 256; i += 256) stage_f32(&h[(size_t)e0*16 + i], &shh[i]);
  for (int i = t; i < 16*d3; i += 256) sy[i] = 0.f;
  stage_fence();
  __syncthreads();

  // A operand (same for every weight tile): 16 edges x 16 hidden, K zero-padded to 32.
  const int r   = lane & 15;
  const int hi2 = lane >> 4;
  const int kb  = hi2 ? 8 : 0;
  v16h a = {};
  for (int j = 0; j < 16; j++) a[j] = hi2 ? (_Float16)0.f : (_Float16)shh[r*16 + j];

  const int np = spec->npaths;
  for (int p = 0; p < np; p++) {
    TPPath P = spec->p[p];
    const int n1 = 2*P.l1+1, n2 = 2*P.l2+1, n3 = 2*P.l3+1;
    const int wsz = P.m1 * P.m2 * P.m3;
    const int nchunks = (wsz + 15) >> 4;

    // ---- weights via WMMA: sw[e][c] = sum_q shh[e][q] * Wt[woff+c][q] ----
    for (int c = wv; c < nchunks; c += 8) {
      const int cc = c*16 + r;                 // weight column within block (N = lane&15)
      v16h b = {};
      if (hi2 == 0 && cc < wsz) {
        const _Float16* wt = Wt + (size_t)(P.woff + cc)*16;   // contiguous 32B row
        __builtin_prefetch(wt + 16*128, 0, 1);                // stream ahead (global_prefetch_b8)
        b = *(const v16h*)wt;
      }
      v8f d = {};
      d = __builtin_amdgcn_wmma_f32_16x16x32_f16(false, a, false, b, (short)0, d, false, false);
      if (cc < wsz) {
        for (int rr = 0; rr < 8; rr++) {
          int e = rr + hi2*8;                  // D row M = rr + 8*hi
          sw[e*wmax + cc] = d[rr];
        }
      }
    }
    __syncthreads();

    // ---- contraction: sy[e][o3 + w*n3 + k] += invfan * sum_{u,v} sw * <x1 x2 C> ----
    const int wk = P.m3 * n3;
    const float* C = w3j + (size_t)(((P.l1*NL) + P.l2)*NL + P.l3) * W3J_STRIDE;
    for (int it = t; it < 16*wk; it += 256) {
      const int e = it / wk, q = it % wk;
      const int w = q / n3, k = q % n3;
      const float* swe  = &sw[e*wmax];
      const float* x1e  = &sx1[e*d1 + P.o1];
      const float* x2e  = &sx2[e*d2 + P.o2];
      float acc = 0.f;
      for (int u = 0; u < P.m1; u++) {
        const float* x1u = x1e + u*n1;
        for (int v = 0; v < P.m2; v++) {
          const float* x2v = x2e + v*n2;
          const float wp = swe[(u*P.m2 + v)*P.m3 + w];
          float s = 0.f;
          for (int i = 0; i < n1; i++) {
            float xv = x1u[i];
            const float* Ci = C + i*81 + k;
            for (int j = 0; j < n2; j++) s += xv * x2v[j] * Ci[j*9];
          }
          acc += wp * s;
        }
      }
      sy[e*d3 + P.o3 + w*n3 + k] += acc * P.invfan;
    }
    __syncthreads();   // protect sw reuse by next path
  }

  // ---- writeback: fused segment-sum (atomics) or dense store ----
  if (dstidx) {
    for (int i = t; i < 16*d3; i += 256) {
      int e = i / d3, c = i % d3;
      atomicAdd(&yout[(size_t)dstidx[e0 + e]*d3 + c], sy[i] * outscale);
    }
  } else {
    for (int i = t; i < 16*d3; i += 256)
      yout[(size_t)e0*d3 + i] = sy[i] * outscale;
  }
}

// ---------------- small helpers ----------------
__global__ void gather_rows_kernel(const float* __restrict__ src, const int* __restrict__ idx,
                                   int rows, int d, float* __restrict__ out){
  int i = blockIdx.x*blockDim.x + threadIdx.x;
  if (i >= rows*d) return;
  int r = i/d, c = i%d;
  out[i] = src[(size_t)idx[r]*d + c];
}
__global__ void compose_idx_kernel(const int* __restrict__ base, const int* __restrict__ ind,
                                   int n, int* __restrict__ out){
  int i = blockIdx.x*blockDim.x + threadIdx.x;
  if (i < n) out[i] = base[ind[i]];
}
__global__ void zero_kernel(float* p, int n){
  int i = blockIdx.x*blockDim.x + threadIdx.x;
  if (i < n) p[i] = 0.f;
}
__global__ void copy_half_kernel(const float* __restrict__ src, int rows, int d, int half,
                                 float* __restrict__ dstp){
  int hd = d/2;
  int i = blockIdx.x*blockDim.x + threadIdx.x;
  if (i >= rows*hd) return;
  int r = i/hd, c = i%hd;
  dstp[i] = src[(size_t)r*d + half*hd + c];
}

// ---------------- host orchestration ----------------
extern "C" void kernel_launch(void* const* d_in, const int* in_sizes, int n_in,
                              void* d_out, int out_size, void* d_ws, size_t ws_size,
                              hipStream_t stream) {
  (void)in_sizes; (void)out_size; (void)ws_size;
  std::vector<TPSpec> hs(6);
  build_tp(IR_IN_, 1, IR_SH_, 3, IR_M1_, 3, &hs[0]);
  build_tp(IR_M1_, 3, IR_SH_, 3, IR_M2_, 6, &hs[1]);
  build_tp(IR_M2_, 6, IR_M2_, 6, IR_M2_, 6, &hs[2]);
  build_tp(IR_M2_, 6, IR_SH_, 3, IR_HH_, 12, &hs[3]);
  build_tp(IR_M2_, 6, IR_SH_, 3, IR_CC_, 68, &hs[4]);
  build_tp(IR_M2_, 6, IR_SH_, 3, IR_CH_, 26, &hs[5]);
  auto ldsb = [&](const TPSpec& s){
    return (size_t)4 * 16 * (s.d1 + s.d2 + 16 + s.d3 + s.wmax);
  };

  bool split = (n_in >= 30);   // 37 inputs when FC tuples are split, 16 when packed
  int ii = 0;
  const float* sh   = (const float*)d_in[ii++];
  const float* emb  = (const float*)d_in[ii++];
  const float* f_in = (const float*)d_in[ii++];
  const float* W[8][4];
  const int nlay[8]  = {3,3,3,4,4,4,4,4};
  const int fnum[8]  = {hs[0].numel, hs[1].numel, hs[2].numel, hs[3].numel,
                        hs[4].numel, hs[5].numel, hs[4].numel, hs[3].numel};
  for (int f = 0; f < 8; ++f) {
    if (split) { for (int l = 0; l < nlay[f]; l++) W[f][l] = (const float*)d_in[ii++]; }
    else {
      const float* base = (const float*)d_in[ii++];
      size_t off = 0;
      int dims[4];
      if (nlay[f] == 3) { dims[0]=16; dims[1]=256; dims[2]=16*fnum[f]; }
      else              { dims[0]=16; dims[1]=256; dims[2]=256; dims[3]=16*fnum[f]; }
      for (int l = 0; l < nlay[f]; l++) { W[f][l] = base + off; off += dims[l]; }
    }
  }
  const int* edge_src = (const int*)d_in[ii++];
  const int* edge_dst = (const int*)d_in[ii++];
  const int* HH_ind   = (const int*)d_in[ii++];
  const int* CC_ind   = (const int*)d_in[ii++];
  const int* CH_ind   = (const int*)d_in[ii++];

  unsigned char* wsb = (unsigned char*)d_ws;
  float*  w3    = (float*)wsb;
  TPSpec* dspec = (TPSpec*)(wsb + SPEC_OFF);
  float*  FB    = (float*)(wsb + BUF_OFF);
  size_t fo = 0;
  auto alloc = [&](size_t n){ float* p = FB + fo; fo += n; return p; };
  float* h_buf = alloc((size_t)N_EDGES*16);
  float* embg  = alloc((size_t)N_BONDS);
  float* nf1   = alloc((size_t)N_NODES*72);
  float* nf2   = alloc((size_t)N_NODES*144);
  float* bondx = alloc((size_t)N_BONDS*144);
  float* bufHH = alloc((size_t)N_BONDS*50);
  float* bufCC = alloc((size_t)N_BONDS*392);
  float* bufCH = alloc((size_t)N_BONDS*140);
  float* nodeC = alloc((size_t)N_NODES*392);
  float* nodeH = alloc((size_t)N_NODES*50);
  int* idxA = (int*)(FB + fo); int* idxB = idxA + N_BONDS; fo += 2*N_BONDS;
  // f16 transposed weight caches (32B-aligned)
  fo = (fo + 15) & ~(size_t)15;
  _Float16* wtbase = (_Float16*)(FB + fo);
  _Float16* wt[8];
  {
    size_t ho = 0;
    for (int f = 0; f < 8; ++f) { wt[f] = wtbase + ho; ho += (size_t)16*fnum[f]; ho = (ho + 15) & ~(size_t)15; }
    fo += (ho + 1) / 2;
  }

  const float inv = 1.0f / sqrtf((float)(N_EDGES / N_NODES));
  auto g1 = [](int n){ return dim3((unsigned)((n + 255)/256)); };

  init_kernel<<<1, 64, 0, stream>>>(wsb);
  for (int f = 0; f < 8; ++f)
    prep_wt_kernel<<<g1(16*fnum[f]), 256, 0, stream>>>(W[f][nlay[f]-1], fnum[f], wt[f]);

  // ---- layer 1: nf1 = segsum( TP1(f_in[src], sh, fc1(emb)) ) * inv ----
  mlp_hidden_kernel<<<N_EDGES/16, 32, 0, stream>>>(emb, W[0][0], W[0][1], nullptr, 1, h_buf);
  zero_kernel<<<g1(N_NODES*72), 256, 0, stream>>>(nf1, N_NODES*72);
  tp16_kernel<<<N_EDGES/16, 256, ldsb(hs[0]), stream>>>(
      f_in, edge_src, sh, nullptr, h_buf, wt[0],
      nf1, edge_dst, inv, &dspec[0], w3);

  // ---- layer 2: nf2 = segsum( TP2(nf1[src], sh, fc2(emb)) ) * inv ----
  mlp_hidden_kernel<<<N_EDGES/16, 32, 0, stream>>>(emb, W[1][0], W[1][1], nullptr, 1, h_buf);
  zero_kernel<<<g1(N_NODES*144), 256, 0, stream>>>(nf2, N_NODES*144);
  tp16_kernel<<<N_EDGES/16, 256, ldsb(hs[1]), stream>>>(
      nf1, edge_src, sh, nullptr, h_buf, wt[1],
      nf2, edge_dst, inv, &dspec[1], w3);

  // ---- bond heads ----
  const int* inds[3] = {HH_ind, CC_ind, CH_ind};
  const int fcx[3]   = {3, 4, 5};
  const int spx[3]   = {3, 4, 5};
  float* outb[3]     = {bufHH, bufCC, bufCH};
  for (int b = 0; b < 3; b++) {
    compose_idx_kernel<<<g1(N_BONDS), 256, 0, stream>>>(edge_src, inds[b], N_BONDS, idxA);
    compose_idx_kernel<<<g1(N_BONDS), 256, 0, stream>>>(edge_dst, inds[b], N_BONDS, idxB);
    gather_rows_kernel<<<g1(N_BONDS), 256, 0, stream>>>(emb, inds[b], N_BONDS, 1, embg);
    mlp_hidden_kernel<<<N_BONDS/16, 32, 0, stream>>>(embg, W[2][0], W[2][1], nullptr, 1, h_buf);
    tp16_kernel<<<N_BONDS/16, 256, ldsb(hs[2]), stream>>>(
        nf2, idxA, nf2, idxB, h_buf, wt[2],
        bondx, nullptr, 1.0f, &dspec[2], w3);
    mlp_hidden_kernel<<<N_BONDS/16, 32, 0, stream>>>(embg, W[fcx[b]][0], W[fcx[b]][1], W[fcx[b]][2], 2, h_buf);
    tp16_kernel<<<N_BONDS/16, 256, ldsb(hs[spx[b]]), stream>>>(
        bondx, nullptr, sh, inds[b], h_buf, wt[fcx[b]],
        outb[b], nullptr, 1.0f, &dspec[spx[b]], w3);
  }

  // ---- node heads: node_C / node_H ----
  mlp_hidden_kernel<<<N_EDGES/16, 32, 0, stream>>>(emb, W[6][0], W[6][1], W[6][2], 2, h_buf);
  zero_kernel<<<g1(N_NODES*392), 256, 0, stream>>>(nodeC, N_NODES*392);
  tp16_kernel<<<N_EDGES/16, 256, ldsb(hs[4]), stream>>>(
      nf2, edge_src, sh, nullptr, h_buf, wt[6],
      nodeC, edge_dst, inv, &dspec[4], w3);

  mlp_hidden_kernel<<<N_EDGES/16, 32, 0, stream>>>(emb, W[7][0], W[7][1], W[7][2], 2, h_buf);
  zero_kernel<<<g1(N_NODES*50), 256, 0, stream>>>(nodeH, N_NODES*50);
  tp16_kernel<<<N_EDGES/16, 256, ldsb(hs[3]), stream>>>(
      nf2, edge_src, sh, nullptr, h_buf, wt[7],
      nodeH, edge_dst, inv, &dspec[3], w3);

  // ---- assemble halves in reference return order ----
  float* out = (float*)d_out;
  struct Part { const float* src; int rows, d; };
  Part parts[5] = { {nodeH, N_NODES, 50}, {nodeC, N_NODES, 392}, {bufHH, N_BONDS, 50},
                    {bufCH, N_BONDS, 140}, {bufCC, N_BONDS, 392} };
  size_t o = 0;
  for (int half = 0; half < 2; half++)
    for (int p = 0; p < 5; p++) {
      int n = parts[p].rows * (parts[p].d/2);
      copy_half_kernel<<<g1(n), 256, 0, stream>>>(parts[p].src, parts[p].rows, parts[p].d, half, out + o);
      o += (size_t)n;
    }
}
